// XNORConv2d_19138374271267
// MI455X (gfx1250) — 8-bit matrix operand order repaired and checked by reading, compiled, not run on hardware
//
#include <hip/hip_runtime.h>
#include <stdint.h>

typedef int   v2i  __attribute__((ext_vector_type(2)));
typedef int   v4i  __attribute__((ext_vector_type(4)));
typedef int   v16i __attribute__((ext_vector_type(16)));
typedef float v8f  __attribute__((ext_vector_type(8)));

#define CIN   256
#define COUT  256
#define HH    56
#define WW    56
#define HWSZ  (HH * WW)
#define NB    32
#define NPIX  (NB * HWSZ)
#define PIXT  64
#define WBUF  (128 * 128)
#define XBUF  (PIXT * 128)

__device__ __forceinline__ unsigned char sign_fp8(float v) {
    return (v > 0.0f) ? 0x38u : ((v < 0.0f) ? 0xB8u : 0x00u);
}

__device__ __forceinline__ unsigned lds_addr32(const void* p) {
    return (unsigned)(uintptr_t)p;
}

__global__ void binarize_x_kernel(const float* __restrict__ x,
                                  unsigned char* __restrict__ xb) {
    int tid = blockIdx.x * blockDim.x + threadIdx.x;
    if (tid >= NPIX * (CIN / 4)) return;
    int pix = tid >> 6;
    int c   = (tid & 63) << 2;
    int n   = pix / HWSZ;
    int hw  = pix - n * HWSZ;
    const float* src = x + ((size_t)(n * CIN + c)) * HWSZ + hw;
    unsigned int packed = 0;
    packed |= (unsigned int)sign_fp8(src[0 * HWSZ]);
    packed |= (unsigned int)sign_fp8(src[1 * HWSZ]) << 8;
    packed |= (unsigned int)sign_fp8(src[2 * HWSZ]) << 16;
    packed |= (unsigned int)sign_fp8(src[3 * HWSZ]) << 24;
    *(unsigned int*)(xb + (size_t)pix * CIN + c) = packed;
}

__global__ void binarize_w_kernel(const float* __restrict__ w,
                                  unsigned char* __restrict__ wb) {
    int tid = blockIdx.x * blockDim.x + threadIdx.x;
    if (tid >= 9 * COUT * CIN) return;
    int c   = tid & (CIN - 1);
    int co  = (tid >> 8) & (COUT - 1);
    int tap = tid >> 16;
    float v = w[((size_t)(co * CIN + c)) * 9 + tap];
    wb[((size_t)(tap * COUT + co)) * CIN + c] = sign_fp8(v);
}

__global__ void __launch_bounds__(256, 2)
xnor_conv_wmma_kernel(const unsigned char* __restrict__ xb,
                      const unsigned char* __restrict__ wb,
                      const float* __restrict__ bias,
                      float* __restrict__ out) {
    __shared__ __align__(16) unsigned char smemW[2][WBUF];
    __shared__ __align__(16) unsigned char smemX[2][XBUF];

    const int tid     = threadIdx.x;
    const int lane    = tid & 31;
    const int wave    = tid >> 5;
    const int pixBase = blockIdx.x * PIXT;
    const int coBase  = blockIdx.y << 7;

    const int aRow  = wave * 16 + (lane & 15);
    const int aHi8  = (lane >> 4) << 4;
    const int bCol  = lane & 15;
    const int bK16  = (lane >> 4) << 4;

    const int sPix = tid >> 2;
    const int sSeg = (tid & 3) << 5;
    const int p    = pixBase + sPix;
    const int pn   = p / HWSZ;
    const int phw  = p - pn * HWSZ;
    const int ph   = phw / WW;
    const int pw   = phw - ph * WW;

    const int wRow = tid >> 1;
    const int wOff = (tid & 1) << 6;

    const unsigned long long wbBase = (unsigned long long)wb;
    const unsigned long long xbBase = (unsigned long long)xb;
    const unsigned ldsWb[2] = { lds_addr32(&smemW[0][wRow * 128 + wOff]),
                                lds_addr32(&smemW[1][wRow * 128 + wOff]) };
    const unsigned ldsXb[2] = { lds_addr32(&smemX[0][sPix * 128 + sSeg]),
                                lds_addr32(&smemX[1][sPix * 128 + sSeg]) };

    auto stage = [&](int chunk, int buf) {
        const int tap  = chunk >> 1;
        const int half = chunk & 1;
        const int kh   = tap / 3;
        const int kw   = tap - kh * 3;
        {
            const unsigned goff =
                (unsigned)((tap * COUT + coBase + wRow) * CIN + half * 128 + wOff);
            asm volatile(
                "global_load_async_to_lds_b128 %0, %1, %2 offset:0\n\t"
                "global_load_async_to_lds_b128 %0, %1, %2 offset:16\n\t"
                "global_load_async_to_lds_b128 %0, %1, %2 offset:32\n\t"
                "global_load_async_to_lds_b128 %0, %1, %2 offset:48"
                :: "v"(ldsWb[buf]), "v"(goff), "s"(wbBase) : "memory");
        }
        {
            const int ih = ph + kh - 1;
            const int iw = pw + kw - 1;
            if ((unsigned)ih < (unsigned)HH && (unsigned)iw < (unsigned)WW) {
                const unsigned goff =
                    (unsigned)((pn * HWSZ + ih * WW + iw) * CIN + half * 128 + sSeg);
                asm volatile(
                    "global_load_async_to_lds_b128 %0, %1, %2 offset:0\n\t"
                    "global_load_async_to_lds_b128 %0, %1, %2 offset:16"
                    :: "v"(ldsXb[buf]), "v"(goff), "s"(xbBase) : "memory");
            } else {
                unsigned char* dst = &smemX[buf][sPix * 128 + sSeg];
                *(v4i*)(dst)      = (v4i){};
                *(v4i*)(dst + 16) = (v4i){};
            }
        }
    };

    v8f acc[4];
    #pragma unroll
    for (int t = 0; t < 4; ++t) acc[t] = (v8f){};

    stage(0, 0);
    asm volatile("s_wait_asynccnt 0x0" ::: "memory");
    __syncthreads();

    for (int chunk = 0; chunk < 18; ++chunk) {
        const int cur = chunk & 1;

        if (chunk + 1 < 18) stage(chunk + 1, cur ^ 1);

        v16i a;
        {
            const unsigned char* base = &smemW[cur][aRow * 128 + aHi8];
            #pragma unroll
            for (int i = 0; i < 4; ++i) {
                v4i t = *(const v4i*)(base + i * 32);
                a[4 * i]     = t.x;
                a[4 * i + 1] = t.y;
                a[4 * i + 2] = t.z;
                a[4 * i + 3] = t.w;
            }
        }

        #pragma unroll
        for (int t = 0; t < 4; ++t) {
            v16i b;
            const unsigned char* base = &smemX[cur][(t * 16 + bCol) * 128 + bK16];
            #pragma unroll
            for (int g = 0; g < 4; ++g) {
                v4i q = *(const v4i*)(base + g * 32);
                b[4 * g]     = q.x;
                b[4 * g + 1] = q.y;
                b[4 * g + 2] = q.z;
                b[4 * g + 3] = q.w;
            }
            acc[t] = __builtin_amdgcn_wmma_f32_16x16x128_fp8_fp8(
                a, b, (short)0, acc[t],  false,  false);
        }

        asm volatile("s_wait_asynccnt 0x0" ::: "memory");
        __syncthreads();
    }

    {
        const int coW = coBase + wave * 16 + ((lane >> 4) << 3);
        #pragma unroll
        for (int t = 0; t < 4; ++t) {
            const int opix = pixBase + t * 16 + (lane & 15);
            const int on   = opix / HWSZ;
            const int ohw  = opix - on * HWSZ;
            #pragma unroll
            for (int r = 0; r < 8; ++r) {
                const int co = coW + r;
                out[((size_t)(on * COUT + co)) * HWSZ + ohw] = acc[t][r] + bias[co];
            }
        }
    }
}

extern "C" void kernel_launch(void* const* d_in, const int* in_sizes, int n_in,
                              void* d_out, int out_size, void* d_ws, size_t ws_size,
                              hipStream_t stream) {
    const float* x    = (const float*)d_in[0];
    const float* w    = (const float*)d_in[1];
    const float* bias = (const float*)d_in[2];
    float* out        = (float*)d_out;

    unsigned char* xb = (unsigned char*)d_ws;
    unsigned char* wb = xb + (size_t)NPIX * CIN;

    {
        int total  = NPIX * (CIN / 4);
        int blocks = (total + 255) / 256;
        binarize_x_kernel<<<blocks, 256, 0, stream>>>(x, xb);
    }
    {
        int total  = 9 * COUT * CIN;
        int blocks = (total + 255) / 256;
        binarize_w_kernel<<<blocks, 256, 0, stream>>>(w, wb);
    }
    {
        dim3 grid(NPIX / PIXT, COUT / 128);
        xnor_conv_wmma_kernel<<<grid, 256, 0, stream>>>(xb, wb, bias, out);
    }
    (void)in_sizes; (void)n_in; (void)out_size; (void)ws_size;
}
